// DecoderLayer_76441827934499
// MI455X (gfx1250) — matrix operand order repaired and checked by reading, not compiled, not run on hardware
//
#include <hip/hip_runtime.h>
#include <hip/hip_bf16.h>

// ---------------------------------------------------------------------------
// MI455X (gfx1250, wave32) decoder layer: all GEMMs on v_wmma_f32_16x16x32_bf16,
// bf16 activation staging via global_load_async_to_lds_b128 (ASYNCcnt).
// BK=64 K-steps, 8 WMMAs per barrier interval, fully unrolled staging.
// ---------------------------------------------------------------------------

typedef __attribute__((ext_vector_type(16))) __bf16   v16bf;
typedef __attribute__((ext_vector_type(8)))  float    v8f;
typedef __attribute__((ext_vector_type(4)))  unsigned uv4;

union Frag { v16bf v; uv4 q[2]; };

constexpr int Bc  = 32;    // batch
constexpr int LDc = 128;   // decoder length
constexpr int LEc = 512;   // encoder length
constexpr int Dc  = 512;   // d_model (== head_dim in this module)
constexpr int Hc  = 8;     // heads
constexpr int HDc = 4096;  // H * D inner width
constexpr int Rc  = 4;     // FFN ratio

constexpr int PITCH = 72;  // LDS row pitch in bf16 halves (64 data + 8 pad)

__device__ __forceinline__ unsigned pack_bf16(float a, float b) {
    union { __bf16 h[2]; unsigned u; } t;
    t.h[0] = (__bf16)a; t.h[1] = (__bf16)b;
    return t.u;
}

// LDS byte offset = low 32 bits of the generic address
__device__ __forceinline__ unsigned lds_off(const void* p) {
    return (unsigned)(unsigned long long)p;
}

// async bf16 16-byte copy: global -> LDS, tracked with ASYNCcnt
__device__ __forceinline__ void async_cp16(const __bf16* gp, const __bf16* lp) {
    unsigned lo = lds_off(lp);
    asm volatile("global_load_async_to_lds_b128 %0, %1, off"
                 :
                 : "v"(lo), "v"(gp)
                 : "memory");
}
__device__ __forceinline__ void async_wait0() {
    asm volatile("s_wait_asynccnt 0x0" ::: "memory");
}

// ---------------------------------------------------------------------------
// Generic weights GEMM: C[M,N] = act(A[M,K] @ W[K,N] + bias)
// A fp32 (converted) or bf16 (async copy); W fp32; C fp32 or bf16.
// Block tile 128x64, BK=64, 256 threads = 8 waves, wave = 16 rows x 64 cols.
// ---------------------------------------------------------------------------
template<bool A_BF16>
__global__ __launch_bounds__(256)
void gemm_wmma(const void* __restrict__ Ain, int lda,
               const float* __restrict__ Bw, int ldb,
               void* __restrict__ Cout, int ldc,
               const float* __restrict__ bias,
               int M, int N, int K, int relu, int out_bf16)
{
    __shared__ __attribute__((aligned(16))) __bf16 sA[128 * PITCH];
    __shared__ __attribute__((aligned(16))) __bf16 sB[64 * PITCH];

    const int tid   = threadIdx.x;
    const int w     = tid >> 5;
    const int lane  = tid & 31;
    const int lhalf = lane >> 4;
    const int l16   = lane & 15;
    const int m0    = blockIdx.y * 128;
    const int n0    = blockIdx.x * 64;

    v8f acc[4] = {};

    for (int kk = 0; kk < K; kk += 64) {
        // ---- stage A tile 128x64 ----
        if (A_BF16) {
            const __bf16* Ab = (const __bf16*)Ain + (size_t)m0 * lda + kk;
#pragma unroll
            for (int it = 0; it < 4; ++it) {
                int i = tid + it * 256;          // 1024 x 16B chunks
                int r = i >> 3, k = (i & 7) * 8;
                async_cp16(&Ab[(size_t)r * lda + k], &sA[r * PITCH + k]);
            }
        } else {
            const float* Af = (const float*)Ain + (size_t)m0 * lda + kk;
#pragma unroll
            for (int it = 0; it < 8; ++it) {
                int i = tid + it * 256;          // 2048 float4s
                int r = i >> 4, k = (i & 15) * 4;
                float4 f = *(const float4*)&Af[(size_t)r * lda + k];
                uint2 s; s.x = pack_bf16(f.x, f.y); s.y = pack_bf16(f.z, f.w);
                *(uint2*)&sA[r * PITCH + k] = s;
            }
        }
        // ---- stage W tile 64x64 -> transposed LDS sB[n][k] ----
        {
            const float* Bp = Bw + (size_t)kk * ldb + n0;
#pragma unroll
            for (int it = 0; it < 4; ++it) {
                int i = tid + it * 256;          // 1024 float4s
                int k = i >> 4, n = (i & 15) * 4;
                float4 f = *(const float4*)&Bp[(size_t)k * ldb + n];
                sB[(n + 0) * PITCH + k] = (__bf16)f.x;
                sB[(n + 1) * PITCH + k] = (__bf16)f.y;
                sB[(n + 2) * PITCH + k] = (__bf16)f.z;
                sB[(n + 3) * PITCH + k] = (__bf16)f.w;
            }
        }
        if (kk + 64 < K)
            __builtin_prefetch((const char*)&Bw[(size_t)(kk + 64) * ldb + n0 + (tid & 63)], 0, 1);
        if (A_BF16) async_wait0();
        __syncthreads();

#pragma unroll
        for (int s = 0; s < 2; ++s) {
            // A fragment: lanes 0-15 hold K{0..7,16..23}, lanes 16-31 K{8..15,24..31}
            Frag af;
            const __bf16* pa = sA + (w * 16 + l16) * PITCH + s * 32 + lhalf * 8;
            af.q[0] = *(const uv4*)(pa);
            af.q[1] = *(const uv4*)(pa + 16);
#pragma unroll
            for (int c = 0; c < 4; ++c) {
                Frag bf;
                const __bf16* pb = sB + (c * 16 + l16) * PITCH + s * 32 + lhalf * 8;
                bf.q[0] = *(const uv4*)(pb);
                bf.q[1] = *(const uv4*)(pb + 16);
                acc[c] = __builtin_amdgcn_wmma_f32_16x16x32_bf16(
                    false, af.v, false, bf.v, (short)0, acc[c], false, false);
            }
        }
        __syncthreads();
    }

#pragma unroll
    for (int c = 0; c < 4; ++c) {
#pragma unroll
        for (int r = 0; r < 8; ++r) {
            int mrow = m0 + w * 16 + r + lhalf * 8;
            int ncol = n0 + c * 16 + l16;
            float v = acc[c][r];
            if (bias) v += bias[ncol];
            if (relu) v = fmaxf(v, 0.f);
            if (out_bf16) ((__bf16*)Cout)[(size_t)mrow * ldc + ncol] = (__bf16)v;
            else          ((float*)Cout)[(size_t)mrow * ldc + ncol] = v;
        }
    }
}

// ---------------------------------------------------------------------------
// Batched attention GEMMs (one block per (b,h) x 64-col tile, M=128 queries)
// SCORES: S[bh,128,Lk] (f32) = Q[bh] (128xD) @ K[bh]^T   (B-op = K rows, async copy)
// CTX:    Ctx[b*LD+q, h*D+n] (bf16) = P[bh] (128xLk) @ V[bh] (V transposed on stage)
// ---------------------------------------------------------------------------
template<bool SCORES>
__global__ __launch_bounds__(256)
void attn_gemm(const __bf16* __restrict__ Aq,
               const __bf16* __restrict__ Bm,
               void* __restrict__ Cout,
               int Lk)
{
    __shared__ __attribute__((aligned(16))) __bf16 sA[128 * PITCH];
    __shared__ __attribute__((aligned(16))) __bf16 sB[64 * PITCH];

    const int tid   = threadIdx.x;
    const int w     = tid >> 5;
    const int lane  = tid & 31;
    const int lhalf = lane >> 4;
    const int l16   = lane & 15;
    const int z  = blockIdx.z, zb = z >> 3, zh = z & 7;  // batch, head
    const int n0 = blockIdx.x * 64;
    const int Ktot = SCORES ? Dc : Lk;

    v8f acc[4] = {};

    for (int kk = 0; kk < Ktot; kk += 64) {
        if (SCORES) {
            // Q rows (row stride HD): async bf16 copy
            const __bf16* Ap = Aq + ((size_t)zb * LDc) * HDc + zh * Dc + kk;
#pragma unroll
            for (int it = 0; it < 4; ++it) {
                int i = tid + it * 256;
                int r = i >> 3, k = (i & 7) * 8;
                async_cp16(&Ap[(size_t)r * HDc + k], &sA[r * PITCH + k]);
            }
            // K rows are the B columns directly: async bf16 copy
            const __bf16* Bp = Bm + ((size_t)zb * Lk + n0) * HDc + zh * Dc + kk;
#pragma unroll
            for (int it = 0; it < 2; ++it) {
                int i = tid + it * 256;
                int n = i >> 3, k = (i & 7) * 8;
                async_cp16(&Bp[(size_t)n * HDc + k], &sB[n * PITCH + k]);
            }
        } else {
            // probs rows (contiguous per bh): async bf16 copy
            const __bf16* Ap = Aq + (size_t)z * LDc * Lk + kk;
#pragma unroll
            for (int it = 0; it < 4; ++it) {
                int i = tid + it * 256;
                int r = i >> 3, k = (i & 7) * 8;
                async_cp16(&Ap[(size_t)r * Lk + k], &sA[r * PITCH + k]);
            }
            // V: b128 global reads, transpose while storing to LDS
            const __bf16* Bp = Bm + ((size_t)zb * Lk + kk) * HDc + zh * Dc + n0;
#pragma unroll
            for (int it = 0; it < 2; ++it) {
                int i = tid + it * 256;
                int k = i >> 3, n = (i & 7) * 8;
                uv4 f = *(const uv4*)&Bp[(size_t)k * HDc + n];
                const __bf16* e = (const __bf16*)&f;
#pragma unroll
                for (int j = 0; j < 8; ++j)
                    sB[(n + j) * PITCH + k] = e[j];
            }
        }
        async_wait0();
        __syncthreads();

#pragma unroll
        for (int s = 0; s < 2; ++s) {
            Frag af;
            const __bf16* pa = sA + (w * 16 + l16) * PITCH + s * 32 + lhalf * 8;
            af.q[0] = *(const uv4*)(pa);
            af.q[1] = *(const uv4*)(pa + 16);
#pragma unroll
            for (int c = 0; c < 4; ++c) {
                Frag bf;
                const __bf16* pb = sB + (c * 16 + l16) * PITCH + s * 32 + lhalf * 8;
                bf.q[0] = *(const uv4*)(pb);
                bf.q[1] = *(const uv4*)(pb + 16);
                acc[c] = __builtin_amdgcn_wmma_f32_16x16x32_bf16(
                    false, af.v, false, bf.v, (short)0, acc[c], false, false);
            }
        }
        __syncthreads();
    }

#pragma unroll
    for (int c = 0; c < 4; ++c) {
#pragma unroll
        for (int r = 0; r < 8; ++r) {
            int mrow = w * 16 + r + lhalf * 8;
            int ncol = n0 + c * 16 + l16;
            if (SCORES)
                ((float*)Cout)[(size_t)z * LDc * Lk + (size_t)mrow * Lk + ncol] = acc[c][r];
            else
                ((__bf16*)Cout)[((size_t)zb * LDc + mrow) * HDc + zh * Dc + ncol] =
                    (__bf16)acc[c][r];
        }
    }
}

// ---------------------------------------------------------------------------
// Row softmax: scale, optional causal mask, f32 scores -> bf16 probs.
// One wave per row (wave32 shuffle reductions), 8 rows per block.
// ---------------------------------------------------------------------------
__global__ __launch_bounds__(256)
void softmax_rows(const float* __restrict__ S, __bf16* __restrict__ P,
                  int Lk, float scale, int causal)
{
    const int w = threadIdx.x >> 5, lane = threadIdx.x & 31;
    const int row = blockIdx.x * 8 + w;
    const int q = row & (LDc - 1);
    const float* src = S + (size_t)row * Lk;

    float vals[16];
    int cnt = 0;
    float mx = -3.0e38f;
    for (int j = lane; j < Lk; j += 32) {
        float v = src[j] * scale;
        if (causal && j > q) v = -3.0e38f;
        vals[cnt++] = v;
        mx = fmaxf(mx, v);
    }
#pragma unroll
    for (int off = 16; off >= 1; off >>= 1) mx = fmaxf(mx, __shfl_xor(mx, off, 32));

    float sum = 0.f;
    for (int i = 0; i < cnt; ++i) { vals[i] = __expf(vals[i] - mx); sum += vals[i]; }
#pragma unroll
    for (int off = 16; off >= 1; off >>= 1) sum += __shfl_xor(sum, off, 32);

    float inv = 1.0f / sum;
    cnt = 0;
    for (int j = lane; j < Lk; j += 32)
        P[(size_t)row * Lk + j] = (__bf16)(vals[cnt++] * inv);
}

// ---------------------------------------------------------------------------
// Out = LayerNorm(X + Res) * g + b   (width 512, one wave per row)
// ---------------------------------------------------------------------------
__global__ __launch_bounds__(256)
void add_layernorm(const float* __restrict__ X, const float* __restrict__ Res,
                   const float* __restrict__ g, const float* __restrict__ b,
                   float* __restrict__ Out)
{
    const int w = threadIdx.x >> 5, lane = threadIdx.x & 31;
    const int row = blockIdx.x * 8 + w;
    const float* px = X + (size_t)row * Dc;
    const float* pr = Res + (size_t)row * Dc;

    float v[16];
    float s = 0.f;
#pragma unroll
    for (int i = 0; i < 16; ++i) {
        int j = lane + i * 32;
        v[i] = px[j] + pr[j];
        s += v[i];
    }
#pragma unroll
    for (int off = 16; off >= 1; off >>= 1) s += __shfl_xor(s, off, 32);
    float mean = s * (1.f / Dc);

    float vs = 0.f;
#pragma unroll
    for (int i = 0; i < 16; ++i) { float d = v[i] - mean; vs += d * d; }
#pragma unroll
    for (int off = 16; off >= 1; off >>= 1) vs += __shfl_xor(vs, off, 32);
    float rstd = rsqrtf(vs * (1.f / Dc) + 1e-5f);

#pragma unroll
    for (int i = 0; i < 16; ++i) {
        int j = lane + i * 32;
        Out[(size_t)row * Dc + j] = (v[i] - mean) * rstd * g[j] + b[j];
    }
}

// ---------------------------------------------------------------------------
extern "C" void kernel_launch(void* const* d_in, const int* in_sizes, int n_in,
                              void* d_out, int out_size, void* d_ws, size_t ws_size,
                              hipStream_t stream)
{
    (void)in_sizes; (void)n_in; (void)out_size; (void)ws_size;
    const float* dec   = (const float*)d_in[0];
    const float* enc   = (const float*)d_in[1];
    // d_in[2] = causal mask; computed analytically (triu, k=1) in softmax.
    const float* sa_wq = (const float*)d_in[3];  const float* sa_bq = (const float*)d_in[4];
    const float* sa_wk = (const float*)d_in[5];  const float* sa_bk = (const float*)d_in[6];
    const float* sa_wv = (const float*)d_in[7];  const float* sa_bv = (const float*)d_in[8];
    const float* sa_wo = (const float*)d_in[9];  const float* sa_bo = (const float*)d_in[10];
    const float* sa_g  = (const float*)d_in[11]; const float* sa_b  = (const float*)d_in[12];
    const float* ca_wq = (const float*)d_in[13]; const float* ca_bq = (const float*)d_in[14];
    const float* ca_wk = (const float*)d_in[15]; const float* ca_bk = (const float*)d_in[16];
    const float* ca_wv = (const float*)d_in[17]; const float* ca_bv = (const float*)d_in[18];
    const float* ca_wo = (const float*)d_in[19]; const float* ca_bo = (const float*)d_in[20];
    const float* ca_g  = (const float*)d_in[21]; const float* ca_b  = (const float*)d_in[22];
    const float* ff_w1 = (const float*)d_in[23]; const float* ff_w2 = (const float*)d_in[24];
    const float* ff_g  = (const float*)d_in[25]; const float* ff_b  = (const float*)d_in[26];

    const int M  = Bc * LDc;   // 4096 decoder rows
    const int ME = Bc * LEc;   // 16384 encoder rows

    char* ws = (char*)d_ws;
    auto alloc = [&](size_t bytes) -> char* {
        char* p = ws;
        ws += (bytes + 255) & ~(size_t)255;
        return p;
    };
    __bf16* Qb  = (__bf16*)alloc((size_t)M  * HDc * 2);
    __bf16* Kb  = (__bf16*)alloc((size_t)ME * HDc * 2);
    __bf16* Vb  = (__bf16*)alloc((size_t)ME * HDc * 2);
    float*  S   = (float*) alloc((size_t)Bc * Hc * LDc * LEc * 4);
    __bf16* P   = (__bf16*)alloc((size_t)Bc * Hc * LDc * LEc * 2);
    __bf16* Ctx = (__bf16*)alloc((size_t)M * HDc * 2);
    float*  T   = (float*) alloc((size_t)M * Dc * 4);
    float*  X1  = (float*) alloc((size_t)M * Dc * 4);
    float*  X2  = (float*) alloc((size_t)M * Dc * 4);
    __bf16* H1  = (__bf16*)alloc((size_t)M * (Dc * Rc) * 2);

    dim3 blk(256);
    const float scale = 0.044194173824159216f;  // 1/sqrt(512)

    // ---------------- self attention ----------------
    gemm_wmma<false><<<dim3(HDc / 64, M / 128), blk, 0, stream>>>(dec, Dc, sa_wq, HDc, Qb, HDc, sa_bq, M, HDc, Dc, 0, 1);
    gemm_wmma<false><<<dim3(HDc / 64, M / 128), blk, 0, stream>>>(dec, Dc, sa_wk, HDc, Kb, HDc, sa_bk, M, HDc, Dc, 0, 1);
    gemm_wmma<false><<<dim3(HDc / 64, M / 128), blk, 0, stream>>>(dec, Dc, sa_wv, HDc, Vb, HDc, sa_bv, M, HDc, Dc, 0, 1);
    attn_gemm<true ><<<dim3(LDc / 64, 1, Bc * Hc), blk, 0, stream>>>(Qb, Kb, S, LDc);
    softmax_rows<<<dim3(Bc * Hc * LDc / 8), blk, 0, stream>>>(S, P, LDc, scale, 1);
    attn_gemm<false><<<dim3(Dc / 64, 1, Bc * Hc), blk, 0, stream>>>(P, Vb, Ctx, LDc);
    gemm_wmma<true ><<<dim3(Dc / 64, M / 128), blk, 0, stream>>>(Ctx, HDc, sa_wo, Dc, T, Dc, sa_bo, M, Dc, HDc, 0, 0);
    add_layernorm<<<dim3(M / 8), blk, 0, stream>>>(T, dec, sa_g, sa_b, X1);

    // ---------------- cross attention ----------------
    gemm_wmma<false><<<dim3(HDc / 64, M / 128), blk, 0, stream>>>(X1, Dc, ca_wq, HDc, Qb, HDc, ca_bq, M, HDc, Dc, 0, 1);
    gemm_wmma<false><<<dim3(HDc / 64, ME / 128), blk, 0, stream>>>(enc, Dc, ca_wk, HDc, Kb, HDc, ca_bk, ME, HDc, Dc, 0, 1);
    gemm_wmma<false><<<dim3(HDc / 64, ME / 128), blk, 0, stream>>>(enc, Dc, ca_wv, HDc, Vb, HDc, ca_bv, ME, HDc, Dc, 0, 1);
    attn_gemm<true ><<<dim3(LEc / 64, 1, Bc * Hc), blk, 0, stream>>>(Qb, Kb, S, LEc);
    softmax_rows<<<dim3(Bc * Hc * LDc / 8), blk, 0, stream>>>(S, P, LEc, scale, 0);
    attn_gemm<false><<<dim3(Dc / 64, 1, Bc * Hc), blk, 0, stream>>>(P, Vb, Ctx, LEc);
    gemm_wmma<true ><<<dim3(Dc / 64, M / 128), blk, 0, stream>>>(Ctx, HDc, ca_wo, Dc, T, Dc, ca_bo, M, Dc, HDc, 0, 0);
    add_layernorm<<<dim3(M / 8), blk, 0, stream>>>(T, X1, ca_g, ca_b, X2);

    // ---------------- feed-forward ----------------
    gemm_wmma<false><<<dim3((Dc * Rc) / 64, M / 128), blk, 0, stream>>>(X2, Dc, ff_w1, Dc * Rc, H1, Dc * Rc, nullptr, M, Dc * Rc, Dc, 1, 1);
    gemm_wmma<true ><<<dim3(Dc / 64, M / 128), blk, 0, stream>>>(H1, Dc * Rc, ff_w2, Dc, T, Dc, nullptr, M, Dc, Dc * Rc, 0, 0);
    add_layernorm<<<dim3(M / 8), blk, 0, stream>>>(T, X2, ff_g, ff_b, (float*)d_out);
}
